// Attention_53730040873051
// MI455X (gfx1250) — compile-verified, matrix operand order checked by reading, not run on hardware
//
#include <hip/hip_runtime.h>

typedef __attribute__((ext_vector_type(16))) _Float16 v16h;
typedef __attribute__((ext_vector_type(8)))  _Float16 v8h;
typedef __attribute__((ext_vector_type(2)))  _Float16 v2h;
typedef __attribute__((ext_vector_type(8)))  float    v8f;
typedef __attribute__((ext_vector_type(4)))  float    v4f;

#define B_   4
#define H_   16
#define S_   2048
#define D_   128
#define QB   128          // q rows per workgroup (8 waves x 16)
#define KB   32           // keys per iteration
#define NWAVE 8
#define NEG_INF (-1e9f)

// LDS strides in halves (all segments 16B aligned; padded against bank conflicts)
#define KSTR 136          // K tile row stride   [key][dim], 272B
#define VSTR 40           // V^T tile row stride [dim][key], 80B
#define PSTR 40           // per-wave P tile     [row][key], 80B

__device__ __forceinline__ v16h cat8(v8h a, v8h b) {
    return __builtin_shufflevector(a, b, 0,1,2,3,4,5,6,7,8,9,10,11,12,13,14,15);
}

__global__ __launch_bounds__(256, 1)
void fa_kernel(const float* __restrict__ Q,
               const float* __restrict__ K,
               const float* __restrict__ V,
               const int*   __restrict__ MASK,
               float*       __restrict__ O)
{
    __shared__ _Float16 ldsK[2][KB * KSTR];       // keys-major, f16
    __shared__ _Float16 ldsV[2][D_ * VSTR];       // dims-major (V transposed), f16
    __shared__ _Float16 ldsP[NWAVE][16 * PSTR];   // per-wave P tile

    const int tid  = threadIdx.x;
    const int lane = tid & 31;
    const int wave = tid >> 5;
    const int ln16 = lane & 15;
    const int hi   = lane >> 4;     // half-wave: selects K-range of A/B operands, row+8 of C
    const int off  = hi * 8;        // K-dim sub-offset within operand layouts

    const int nqb = S_ / QB;        // 16
    const int bh  = blockIdx.x / nqb;
    const int qb  = blockIdx.x % nqb;
    const int b   = bh / H_;
    const size_t base = (size_t)bh * S_ * D_;
    const int q0  = qb * QB + wave * 16;

    // all-ones B operand: one extra WMMA turns P*ones into per-row sums
    v16h ones;
#pragma unroll
    for (int i = 0; i < 16; ++i) ones[i] = (_Float16)1.0f;

    // ---------- load this wave's 16x128 Q tile into A-layout f16 registers ----------
    v16h qa[4];
    {
        const float* qrow = Q + base + (size_t)(q0 + ln16) * D_;
#pragma unroll
        for (int kb = 0; kb < 4; ++kb) {
            const float* p0 = qrow + kb * 32 + off;
            v4f f0 = *(const v4f*)(p0 + 0);
            v4f f1 = *(const v4f*)(p0 + 4);
            v4f f2 = *(const v4f*)(p0 + 16);
            v4f f3 = *(const v4f*)(p0 + 20);
            v16h a;
#pragma unroll
            for (int j = 0; j < 4; ++j) {
                a[j]      = (_Float16)f0[j];
                a[4 + j]  = (_Float16)f1[j];
                a[8 + j]  = (_Float16)f2[j];
                a[12 + j] = (_Float16)f3[j];
            }
            qa[kb] = a;
        }
    }

    // ---------- accumulators / softmax state ----------
    v8f o_acc[8];
#pragma unroll
    for (int nt = 0; nt < 8; ++nt) o_acc[nt] = v8f{};
    float m_run[8], l_run[8];
#pragma unroll
    for (int r = 0; r < 8; ++r) { m_run[r] = -INFINITY; l_run[r] = 0.f; }

    // ---------- tile staging: thread owns 2 keys x 8 dims of K and V ----------
    const int srow = (tid & 15) * 2;    // key pair: srow, srow+1
    const int scol = (tid >> 4) * 8;    // dim segment start
    v4f kreg[4], vreg[4];               // fp32 staging (dies at convert_stage)
    v8h kh0, kh1;                       // f16 staging (lives to store_tile)
    v2h vh[8];

    auto load_tile = [&](int it) {
        const float* kp = K + base + (size_t)(it * KB + srow) * D_ + scol;
        const float* vp = V + base + (size_t)(it * KB + srow) * D_ + scol;
        kreg[0] = *(const v4f*)(kp);            kreg[1] = *(const v4f*)(kp + 4);
        kreg[2] = *(const v4f*)(kp + D_);       kreg[3] = *(const v4f*)(kp + D_ + 4);
        vreg[0] = *(const v4f*)(vp);            vreg[1] = *(const v4f*)(vp + 4);
        vreg[2] = *(const v4f*)(vp + D_);       vreg[3] = *(const v4f*)(vp + D_ + 4);
    };
    auto convert_stage = [&]() {
#pragma unroll
        for (int e = 0; e < 4; ++e) {
            kh0[e] = (_Float16)kreg[0][e];  kh0[4 + e] = (_Float16)kreg[1][e];
            kh1[e] = (_Float16)kreg[2][e];  kh1[4 + e] = (_Float16)kreg[3][e];
        }
#pragma unroll
        for (int d = 0; d < 8; ++d) {
            vh[d][0] = (_Float16)((d < 4) ? vreg[0][d] : vreg[1][d - 4]);
            vh[d][1] = (_Float16)((d < 4) ? vreg[2][d] : vreg[3][d - 4]);
        }
    };
    auto store_tile = [&](int buf) {
        _Float16* kt = ldsK[buf];
        _Float16* vt = ldsV[buf];
        *(v8h*)(kt + srow * KSTR + scol)       = kh0;
        *(v8h*)(kt + (srow + 1) * KSTR + scol) = kh1;
#pragma unroll
        for (int d = 0; d < 8; ++d)
            *(v2h*)(vt + (scol + d) * VSTR + srow) = vh[d];
    };

    load_tile(0);
    convert_stage();
    store_tile(0);
    __syncthreads();

    const int niter = S_ / KB;          // 64
    for (int it = 0; it < niter; ++it) {
        const int buf = it & 1;

        // software pipeline: fetch next tile into registers while computing
        if (it + 1 < niter) load_tile(it + 1);

        // ---------- scores: C = Q * K^T, depth-2 pipelined B operands ----------
        const _Float16* kt = ldsK[buf];
        auto loadB = [&](int sub, int kb) -> v16h {
            const _Float16* p = kt + (sub * 16 + ln16) * KSTR + kb * 32 + off;
            return cat8(*(const v8h*)(p), *(const v8h*)(p + 16));
        };
        v8f c0 = v8f{}, c1 = v8f{};
        {
            v16h b00 = loadB(0, 0), b10 = loadB(1, 0);
            v16h b01 = loadB(0, 1), b11 = loadB(1, 1);
            // kb = 0
            c0 = __builtin_amdgcn_wmma_f32_16x16x32_f16(false, qa[0], false, b00, (short)0, c0, false, false);
            c1 = __builtin_amdgcn_wmma_f32_16x16x32_f16(false, qa[0], false, b10, (short)0, c1, false, false);
            b00 = loadB(0, 2); b10 = loadB(1, 2);
            // kb = 1
            c0 = __builtin_amdgcn_wmma_f32_16x16x32_f16(false, qa[1], false, b01, (short)0, c0, false, false);
            c1 = __builtin_amdgcn_wmma_f32_16x16x32_f16(false, qa[1], false, b11, (short)0, c1, false, false);
            b01 = loadB(0, 3); b11 = loadB(1, 3);
            // kb = 2
            c0 = __builtin_amdgcn_wmma_f32_16x16x32_f16(false, qa[2], false, b00, (short)0, c0, false, false);
            c1 = __builtin_amdgcn_wmma_f32_16x16x32_f16(false, qa[2], false, b10, (short)0, c1, false, false);
            // kb = 3
            c0 = __builtin_amdgcn_wmma_f32_16x16x32_f16(false, qa[3], false, b01, (short)0, c0, false, false);
            c1 = __builtin_amdgcn_wmma_f32_16x16x32_f16(false, qa[3], false, b11, (short)0, c1, false, false);
        }

        // ---------- mask (per key, matches reference -1e9 fill) ----------
        const int kbase = it * KB;
        const int mk0 = MASK[b * S_ + kbase + ln16];
        const int mk1 = MASK[b * S_ + kbase + 16 + ln16];
        if (mk0 == 0) {
#pragma unroll
            for (int r = 0; r < 8; ++r) c0[r] = NEG_INF;
        }
        if (mk1 == 0) {
#pragma unroll
            for (int r = 0; r < 8; ++r) c1[r] = NEG_INF;
        }

        // ---------- online softmax; emit P (f16) to per-wave LDS ----------
        _Float16* pl = ldsP[wave];
        float scl[8];
#pragma unroll
        for (int r = 0; r < 8; ++r) {
            float rm = fmaxf(c0[r], c1[r]);
#pragma unroll
            for (int s = 8; s >= 1; s >>= 1) rm = fmaxf(rm, __shfl_xor(rm, s, 32));
            const float mn = fmaxf(m_run[r], rm);
            scl[r] = __expf(m_run[r] - mn);
            m_run[r] = mn;
            const float p0 = __expf(c0[r] - mn);
            const float p1 = __expf(c1[r] - mn);
            const int m = r + 8 * hi;                    // C-layout row
            pl[m * PSTR + ln16]      = (_Float16)p0;
            pl[m * PSTR + 16 + ln16] = (_Float16)p1;
        }
#pragma unroll
        for (int nt = 0; nt < 8; ++nt) {
#pragma unroll
            for (int r = 0; r < 8; ++r) o_acc[nt][r] *= scl[r];
        }

        // fp32 staging regs die here; f16 staging (16 VGPRs) lives to store_tile
        if (it + 1 < niter) convert_stage();

        // ---------- read P back in A layout ----------
        v16h pa;
        {
            const _Float16* p = pl + ln16 * PSTR + off;
            pa = cat8(*(const v8h*)(p), *(const v8h*)(p + 16));
        }

        // row sums of P via WMMA against all-ones B (every column of C = row sum)
        v8f lsum = __builtin_amdgcn_wmma_f32_16x16x32_f16(
                       false, pa, false, ones, (short)0, v8f{}, false, false);
#pragma unroll
        for (int r = 0; r < 8; ++r) l_run[r] = l_run[r] * scl[r] + lsum[r];

        // ---------- O += P * V, depth-2 pipelined V operands ----------
        const _Float16* vt = ldsV[buf];
        auto loadV = [&](int nt) -> v16h {
            const _Float16* p = vt + (nt * 16 + ln16) * VSTR + off;
            return cat8(*(const v8h*)(p), *(const v8h*)(p + 16));
        };
        {
            v16h va = loadV(0), vb = loadV(1);
            o_acc[0] = __builtin_amdgcn_wmma_f32_16x16x32_f16(false, pa, false, va, (short)0, o_acc[0], false, false);
            va = loadV(2);
            o_acc[1] = __builtin_amdgcn_wmma_f32_16x16x32_f16(false, pa, false, vb, (short)0, o_acc[1], false, false);
            vb = loadV(3);
            o_acc[2] = __builtin_amdgcn_wmma_f32_16x16x32_f16(false, pa, false, va, (short)0, o_acc[2], false, false);
            va = loadV(4);
            o_acc[3] = __builtin_amdgcn_wmma_f32_16x16x32_f16(false, pa, false, vb, (short)0, o_acc[3], false, false);
            vb = loadV(5);
            o_acc[4] = __builtin_amdgcn_wmma_f32_16x16x32_f16(false, pa, false, va, (short)0, o_acc[4], false, false);
            va = loadV(6);
            o_acc[5] = __builtin_amdgcn_wmma_f32_16x16x32_f16(false, pa, false, vb, (short)0, o_acc[5], false, false);
            vb = loadV(7);
            o_acc[6] = __builtin_amdgcn_wmma_f32_16x16x32_f16(false, pa, false, va, (short)0, o_acc[6], false, false);
            o_acc[7] = __builtin_amdgcn_wmma_f32_16x16x32_f16(false, pa, false, vb, (short)0, o_acc[7], false, false);
        }

        __syncthreads();                   // all waves done reading current buffers
        if (it + 1 < niter) store_tile(buf ^ 1);
        __syncthreads();                   // next tile visible
    }

    // ---------- normalize and store fp32 output ----------
#pragma unroll
    for (int r = 0; r < 8; ++r) {
        const int qrow = q0 + r + 8 * hi;
        const float inv = 1.f / l_run[r];
        float* orow = O + base + (size_t)qrow * D_;
#pragma unroll
        for (int nt = 0; nt < 8; ++nt)
            orow[nt * 16 + ln16] = o_acc[nt][r] * inv;
    }
}

extern "C" void kernel_launch(void* const* d_in, const int* in_sizes, int n_in,
                              void* d_out, int out_size, void* d_ws, size_t ws_size,
                              hipStream_t stream) {
    (void)in_sizes; (void)n_in; (void)out_size; (void)d_ws; (void)ws_size;
    const float* Q    = (const float*)d_in[0];
    const float* K    = (const float*)d_in[1];
    const float* V    = (const float*)d_in[2];
    const int*   MASK = (const int*)d_in[3];
    float*       O    = (float*)d_out;

    dim3 grid(B_ * H_ * (S_ / QB));   // 1024 workgroups
    dim3 block(256);                  // 8 waves (wave32)
    fa_kernel<<<grid, block, 0, stream>>>(Q, K, V, MASK, O);
}
